// RobertaSelfAttention_34024730919581
// MI455X (gfx1250) — matrix operand order repaired and checked by reading, not compiled, not run on hardware
//
#include <hip/hip_runtime.h>
#include <hip/hip_bf16.h>

typedef __bf16 bf16_t;
typedef __attribute__((ext_vector_type(16))) __bf16 v16bf;
typedef __attribute__((ext_vector_type(8)))  __bf16 v8bf;
typedef __attribute__((ext_vector_type(8)))  float  v8f;

constexpr int NB = 4;      // batch
constexpr int L  = 1024;   // seq len
constexpr int DM = 1024;   // model dim
constexpr int NH = 16;     // heads
constexpr int HD = 64;     // head dim
constexpr int BH = NB * NH;

// ---------- helpers ----------

__device__ __forceinline__ bf16_t f2bf(float f) {
  unsigned u = __builtin_bit_cast(unsigned, f);
  unsigned r = u + 0x7fffu + ((u >> 16) & 1u);   // round-to-nearest-even
  unsigned short hs = (unsigned short)(r >> 16);
  return __builtin_bit_cast(bf16_t, hs);
}

__device__ __forceinline__ v16bf cat8(v8bf lo, v8bf hi) {
  return __builtin_shufflevector(lo, hi, 0,1,2,3,4,5,6,7,8,9,10,11,12,13,14,15);
}

// A-fragment (16x32 bf16, M x K): lane m = lane&15 selects row, h = lane>>4.
// Per ISA 7.12.2: lane holds K = d0+8h+[0,8) and d0+16+8h+[0,8)  (two contig runs)
__device__ __forceinline__ v16bf loadA16(const bf16_t* p, int ld, int m, int h, int d0) {
  const bf16_t* q = p + (size_t)m * ld + d0 + 8 * h;
  return cat8(*(const v8bf*)q, *(const v8bf*)(q + 16));
}

// B-fragment (32x16 bf16, K x N): lane n = lane&15 selects column (a source row
// of the K-contiguous operand), lane holds K = d0+8h+[0,8) and d0+16+8h+[0,8).
__device__ __forceinline__ v16bf loadB16(const bf16_t* p, int ld, int n, int h, int d0) {
  const bf16_t* q = p + (size_t)n * ld + d0 + 8 * h;
  return cat8(*(const v8bf*)q, *(const v8bf*)(q + 16));
}

__device__ __forceinline__ v8f wmma_bf(v16bf a, v16bf b, v8f c) {
  return __builtin_amdgcn_wmma_f32_16x16x32_bf16(false, a, false, b, (short)0, c,
                                                 false, false);
}

// ---------- kernel A: fp32 -> bf16 prepack ----------

__global__ void cvt_f32_bf16(const float* __restrict__ s, bf16_t* __restrict__ d, int n) {
  int i = blockIdx.x * 256 + threadIdx.x;
  if (i < n) d[i] = f2bf(s[i]);
}

// ---------- kernel B: QKV projection (C = X * W^T + b) ----------
// Wave = 32x64 output tile, k-loop double-buffered (ping-pong fragment sets).
// grid.x = 256 (8 waves/block, 2048 tiles/z), grid.y = z in {Q,K,V}.

__device__ __forceinline__ void proj_load_chunk(const bf16_t* hb, const bf16_t* w,
                                                int i0, int o0, int n16, int hh,
                                                int k0, v16bf A[2], v16bf Bf[4]) {
  A[0] = loadA16(hb + (size_t)i0 * DM, DM, n16, hh, k0);
  A[1] = loadA16(hb + (size_t)(i0 + 16) * DM, DM, n16, hh, k0);
#pragma unroll
  for (int nt = 0; nt < 4; nt++)
    Bf[nt] = loadB16(w + (size_t)(o0 + 16 * nt) * DM, DM, n16, hh, k0);
}

__device__ __forceinline__ void proj_mm(v8f acc[2][4], const v16bf A[2],
                                        const v16bf Bf[4]) {
#pragma unroll
  for (int nt = 0; nt < 4; nt++)
#pragma unroll
    for (int mt = 0; mt < 2; mt++)
      acc[mt][nt] = wmma_bf(A[mt], Bf[nt], acc[mt][nt]);
}

__global__ __launch_bounds__(256) void proj_kernel(
    const bf16_t* __restrict__ hb,
    const bf16_t* __restrict__ wqb, const bf16_t* __restrict__ wkb,
    const bf16_t* __restrict__ wvb,
    const float* __restrict__ bq, const float* __restrict__ bk,
    const float* __restrict__ bv,
    bf16_t* __restrict__ qb, bf16_t* __restrict__ kb, bf16_t* __restrict__ vt) {
  const int z    = blockIdx.y;
  const int lane = threadIdx.x & 31, wid = threadIdx.x >> 5;
  const int n16  = lane & 15, hh = lane >> 4;
  const int gw   = blockIdx.x * 8 + wid;           // 2048 tiles
  const int i0   = (gw >> 4) * 32;                 // output row block (32 rows)
  const int o0   = (gw & 15) * 64;                 // output col block (one head)
  const bf16_t* w    = (z == 0) ? wqb : (z == 1) ? wkb : wvb;
  const float*  bias = (z == 0) ? bq  : (z == 1) ? bk  : bv;

  const v8f zero8 = {0,0,0,0,0,0,0,0};
  v8f acc[2][4];
#pragma unroll
  for (int mt = 0; mt < 2; mt++)
#pragma unroll
    for (int nt = 0; nt < 4; nt++) acc[mt][nt] = zero8;

  v16bf A0[2], B0[4], A1[2], B1[4];
  proj_load_chunk(hb, w, i0, o0, n16, hh, 0, A0, B0);
  for (int k0 = 0; k0 < DM; k0 += 64) {
    proj_load_chunk(hb, w, i0, o0, n16, hh, k0 + 32, A1, B1);
    proj_mm(acc, A0, B0);
    if (k0 + 64 < DM) proj_load_chunk(hb, w, i0, o0, n16, hh, k0 + 64, A0, B0);
    proj_mm(acc, A1, B1);
  }

  const int bidx = i0 >> 10, l = i0 & 1023, hsel = o0 >> 6;
#pragma unroll
  for (int mt = 0; mt < 2; mt++)
#pragma unroll
    for (int nt = 0; nt < 4; nt++) {
      float bval = bias[o0 + 16 * nt + n16];
#pragma unroll
      for (int v = 0; v < 8; v++) {
        float val = acc[mt][nt][v] + bval;
        int row = l + 16 * mt + v + 8 * hh;        // D-frag: m = v + 8*(lane>>4)
        int dl  = 16 * nt + n16;
        if (z == 2) {                               // V transposed: [bh][d][r]
          vt[((size_t)((bidx * NH + hsel) * HD + dl)) * L + row] = f2bf(val);
        } else {                                    // Q/K head-major: [bh][l][d]
          bf16_t* dst = (z == 0) ? qb : kb;
          dst[((size_t)((bidx * NH + hsel) * L + row)) * HD + dl] = f2bf(val);
        }
      }
    }
}

// ---------- kernel C: fused relative-position flash attention ----------
// One wave = one (b,h) and 16 query rows. 4 waves/block, grid.x = 1024.
// Per-wave LDS: 2x term1 helper [16][32]f32, 2x term2 helper [32][16]f32,
// probs [16][32]bf16  = 9216 B.

__global__ __launch_bounds__(128) void attn_kernel(
    const bf16_t* __restrict__ qb, const bf16_t* __restrict__ kb,
    const bf16_t* __restrict__ vt, const bf16_t* __restrict__ eb,
    const float* __restrict__ mask, float* __restrict__ out) {
  const int lane = threadIdx.x & 31, wid = threadIdx.x >> 5;
  const int n16  = lane & 15, hh = lane >> 4;
  const int gw   = blockIdx.x * 4 + wid;           // 4096 waves total
  const int bh   = gw >> 6;
  const int l0   = (gw & 63) * 16;
  const int bidx = bh >> 4, head = bh & 15;

  __shared__ char smem_raw[4 * 9216];
  char*   my  = smem_raw + wid * 9216;
  float*  Ml[2]; float* Nl[2];
  Ml[0] = (float*)my;            // [16][32]
  Ml[1] = Ml[0] + 512;
  Nl[0] = Ml[1] + 512;           // [32][16]
  Nl[1] = Nl[0] + 512;
  bf16_t* Pl = (bf16_t*)(Nl[1] + 512);   // [16][32]

  const bf16_t* qrow = qb + ((size_t)bh * L + l0) * HD;
  v16bf Qa0 = loadA16(qrow, HD, n16, hh, 0);
  v16bf Qa1 = loadA16(qrow, HD, n16, hh, 32);

  const v8f zero8 = {0,0,0,0,0,0,0,0};
  v8f O0 = zero8, O1 = zero8, O2 = zero8, O3 = zero8;
  float mrow[8], ssum[8];
#pragma unroll
  for (int v = 0; v < 8; v++) { mrow[v] = -3.0e38f; ssum[v] = 0.0f; }

  for (int r0 = 0; r0 < L; r0 += 32) {
    // ---- V fragments for this chunk: issue first, consumed last ----
    const bf16_t* vbp = vt + (size_t)(bh * HD + n16) * L + r0 + 16 * hh;
    v16bf Vb0 = *(const v16bf*)(vbp);
    v16bf Vb1 = *(const v16bf*)(vbp + 16 * L);
    v16bf Vb2 = *(const v16bf*)(vbp + 32 * L);
    v16bf Vb3 = *(const v16bf*)(vbp + 48 * L);

    // prefetch next chunk's K tile (32 rows x 64 bf16 = 32 cachelines)
    if (r0 + 32 < L)
      __builtin_prefetch(kb + ((size_t)bh * L + r0 + 32) * HD + lane * 64, 0, 0);

    // ---- score WMMA stages for both 16-col tiles, separate LDS sets ----
    v8f s[2];
#pragma unroll
    for (int t = 0; t < 2; t++) {
      const int r = r0 + 16 * t;
      const bf16_t* krow = kb + ((size_t)bh * L + r) * HD;
      v16bf Kb0 = loadB16(krow, HD, n16, hh, 0);
      v16bf Kb1 = loadB16(krow, HD, n16, hh, 32);

      // term0: Q x K^T
      v8f sc = zero8;
      sc = wmma_bf(Qa0, Kb0, sc);
      sc = wmma_bf(Qa1, Kb1, sc);
      s[t] = sc;

      // 32-row dist_emb band covering l-r+1023 for this tile (in [0,2047])
      const int base = l0 - r + 1008;
      const bf16_t* e0 = eb + (size_t)base * HD;
      const bf16_t* e1 = e0 + 16 * HD;

      // term2 helper: N[u,rr] = e[base+u] . k[r+rr]
      v8f n0 = zero8, n1 = zero8;
      n0 = wmma_bf(loadA16(e0, HD, n16, hh, 0),  Kb0, n0);
      n0 = wmma_bf(loadA16(e0, HD, n16, hh, 32), Kb1, n0);
      n1 = wmma_bf(loadA16(e1, HD, n16, hh, 0),  Kb0, n1);
      n1 = wmma_bf(loadA16(e1, HD, n16, hh, 32), Kb1, n1);

      // term1 helper: M[i,u] = q[l0+i] . e[base+u]
      v8f m0 = zero8, m1 = zero8;
      m0 = wmma_bf(Qa0, loadB16(e0, HD, n16, hh, 0),  m0);
      m0 = wmma_bf(Qa1, loadB16(e0, HD, n16, hh, 32), m0);
      m1 = wmma_bf(Qa0, loadB16(e1, HD, n16, hh, 0),  m1);
      m1 = wmma_bf(Qa1, loadB16(e1, HD, n16, hh, 32), m1);

#pragma unroll
      for (int v = 0; v < 8; v++) {
        int u = v + 8 * hh;                        // D-frag row
        Nl[t][u * 16 + n16]        = n0[v];
        Nl[t][(16 + u) * 16 + n16] = n1[v];
        Ml[t][u * 32 + n16]        = m0[v];
        Ml[t][u * 32 + 16 + n16]   = m1[v];
      }
    }
    asm volatile("s_wait_dscnt 0" ::: "memory");

    // ---- diagonal gathers + mask for both tiles ----
    float sv0[8], sv1[8];
    float mk0 = mask[(size_t)bidx * L + r0 + n16];
    float mk1 = mask[(size_t)bidx * L + r0 + 16 + n16];
#pragma unroll
    for (int v = 0; v < 8; v++) {
      int i = v + 8 * hh;
      int u = i - n16 + 15;                        // diagonal gather index
      sv0[v] = (s[0][v] + Ml[0][i * 32 + u] + Nl[0][u * 16 + n16]) * 0.125f + mk0;
      sv1[v] = (s[1][v] + Ml[1][i * 32 + u] + Nl[1][u * 16 + n16]) * 0.125f + mk1;
    }

    // ---- online softmax over this 32-column chunk (row = 16-lane group) ----
#pragma unroll
    for (int v = 0; v < 8; v++) {
      float a = fmaxf(sv0[v], sv1[v]);
      a = fmaxf(a, __shfl_xor(a, 1, 16));
      a = fmaxf(a, __shfl_xor(a, 2, 16));
      a = fmaxf(a, __shfl_xor(a, 4, 16));
      a = fmaxf(a, __shfl_xor(a, 8, 16));
      float nm    = fmaxf(mrow[v], a);
      float alpha = __expf(mrow[v] - nm);
      mrow[v] = nm;
      float p0 = __expf(sv0[v] - nm);
      float p1 = __expf(sv1[v] - nm);
      float rs = p0 + p1;
      rs += __shfl_xor(rs, 1, 16);
      rs += __shfl_xor(rs, 2, 16);
      rs += __shfl_xor(rs, 4, 16);
      rs += __shfl_xor(rs, 8, 16);
      ssum[v] = ssum[v] * alpha + rs;
      int i = v + 8 * hh;
      Pl[i * 32 + n16]      = f2bf(p0);
      Pl[i * 32 + 16 + n16] = f2bf(p1);
      O0[v] *= alpha; O1[v] *= alpha; O2[v] *= alpha; O3[v] *= alpha;
    }
    asm volatile("s_wait_dscnt 0" ::: "memory");

    // ---- PV: probs(16x32) x V(32x64), V pre-transposed so B-frags contiguous ----
    const bf16_t* pl = Pl + n16 * 32 + 16 * hh;
    v16bf Pa = *(const v16bf*)pl;
    O0 = wmma_bf(Pa, Vb0, O0);
    O1 = wmma_bf(Pa, Vb1, O1);
    O2 = wmma_bf(Pa, Vb2, O2);
    O3 = wmma_bf(Pa, Vb3, O3);
  }

  // normalize and store ctx in [B, L, H*HD] layout
#pragma unroll
  for (int v = 0; v < 8; v++) {
    float inv = 1.0f / ssum[v];
    int row = l0 + v + 8 * hh;
    float* op = out + ((size_t)(bidx * L + row)) * DM + head * HD;
    op[n16]      = O0[v] * inv;
    op[16 + n16] = O1[v] * inv;
    op[32 + n16] = O2[v] * inv;
    op[48 + n16] = O3[v] * inv;
  }
}

// ---------- host ----------

extern "C" void kernel_launch(void* const* d_in, const int* in_sizes, int n_in,
                              void* d_out, int out_size, void* d_ws, size_t ws_size,
                              hipStream_t stream) {
  (void)in_sizes; (void)n_in; (void)out_size; (void)ws_size;
  const float* hidden = (const float*)d_in[0];
  const float* amask  = (const float*)d_in[1];
  const float* Wq = (const float*)d_in[2];
  const float* bq = (const float*)d_in[3];
  const float* Wk = (const float*)d_in[4];
  const float* bk = (const float*)d_in[5];
  const float* Wv = (const float*)d_in[6];
  const float* bv = (const float*)d_in[7];
  const float* de = (const float*)d_in[8];
  float* out = (float*)d_out;

  char* ws = (char*)d_ws;
  size_t off = 0;
  bf16_t* hb  = (bf16_t*)(ws + off); off += (size_t)NB * L * DM * 2;   // 8 MB
  bf16_t* wqb = (bf16_t*)(ws + off); off += (size_t)DM * DM * 2;       // 2 MB
  bf16_t* wkb = (bf16_t*)(ws + off); off += (size_t)DM * DM * 2;
  bf16_t* wvb = (bf16_t*)(ws + off); off += (size_t)DM * DM * 2;
  bf16_t* eb  = (bf16_t*)(ws + off); off += (size_t)2048 * HD * 2;     // padded
  bf16_t* qbf = (bf16_t*)(ws + off); off += (size_t)BH * L * HD * 2;   // 8 MB
  bf16_t* kbf = (bf16_t*)(ws + off); off += (size_t)BH * L * HD * 2;
  bf16_t* vtb = (bf16_t*)(ws + off); off += (size_t)BH * HD * L * 2;

  const int nh = NB * L * DM;     // 4194304
  const int nw = DM * DM;         // 1048576
  const int ne = 2047 * HD;       // 131008
  cvt_f32_bf16<<<(nh + 255) / 256, 256, 0, stream>>>(hidden, hb, nh);
  cvt_f32_bf16<<<(nw + 255) / 256, 256, 0, stream>>>(Wq, wqb, nw);
  cvt_f32_bf16<<<(nw + 255) / 256, 256, 0, stream>>>(Wk, wkb, nw);
  cvt_f32_bf16<<<(nw + 255) / 256, 256, 0, stream>>>(Wv, wvb, nw);
  cvt_f32_bf16<<<(ne + 255) / 256, 256, 0, stream>>>(de, eb, ne);
  hipMemsetAsync(eb + (size_t)2047 * HD, 0, HD * sizeof(bf16_t), stream); // pad row

  proj_kernel<<<dim3(256, 3, 1), 256, 0, stream>>>(hb, wqb, wkb, wvb, bq, bk, bv,
                                                   qbf, kbf, vtb);
  attn_kernel<<<dim3(1024, 1, 1), 128, 0, stream>>>(qbf, kbf, vtb, eb, amask, out);
}
